// BiLSTM_7928509628689
// MI455X (gfx1250) — compile-verified, matrix operand order checked by reading, not run on hardware
//
#include <hip/hip_runtime.h>

#define DEV __device__ __forceinline__

typedef __attribute__((ext_vector_type(16))) __bf16 v16bf;
typedef __attribute__((ext_vector_type(8)))  float  v8f;
typedef unsigned int  u32x4 __attribute__((ext_vector_type(4)));
typedef int           i32x8 __attribute__((ext_vector_type(8)));
typedef int           i32x4 __attribute__((ext_vector_type(4)));

constexpr int N_SEQ  = 2048;
constexpr int T_LEN  = 64;
constexpr int D_IN   = 512;
constexpr int H_DIM  = 256;
constexpr int G_DIM  = 4 * H_DIM;            // 1024
constexpr int TOKENS = N_SEQ * T_LEN;        // 131072

#if defined(__has_builtin)
#if __has_builtin(__builtin_amdgcn_tensor_load_to_lds) && __has_builtin(__builtin_amdgcn_s_wait_tensorcnt)
#define HAVE_TDM 1
#endif
#if __has_builtin(__builtin_amdgcn_perm)
#define HAVE_PERM 1
#endif
#endif

DEV unsigned short f2bf(float f) {            // RNE (used on cold paths)
    unsigned u = __builtin_bit_cast(unsigned, f);
    u += 0x7FFFu + ((u >> 16) & 1u);
    return (unsigned short)(u >> 16);
}
DEV float bf2f(unsigned short v) {
    return __builtin_bit_cast(float, (unsigned)v << 16);
}
DEV unsigned pk2bf(float lo, float hi) {      // pack two bf16 (truncate) in 1 VALU op
#ifdef HAVE_PERM
    return __builtin_amdgcn_perm(__builtin_bit_cast(unsigned, hi),
                                 __builtin_bit_cast(unsigned, lo), 0x07060302u);
#else
    return ((unsigned)f2bf(hi) << 16) | (unsigned)f2bf(lo);
#endif
}

// ---------------- phase 0: weights fp32 -> bf16 (row-major (4H,K) kept) ----------------
__global__ __launch_bounds__(256) void cvt_f32_bf16(const float* __restrict__ src,
                                                    unsigned short* __restrict__ dst, int n) {
    int i = blockIdx.x * blockDim.x + threadIdx.x;
    int stride = gridDim.x * blockDim.x;
    for (; i < n; i += stride) dst[i] = f2bf(src[i]);
}

#ifdef HAVE_TDM
// One 2D TDM descriptor: copy a 128-row x 32-element bf16 tile (row stride 512 elems)
// from global into LDS, rows packed contiguously (128*32 bf16) -> exactly our sb layout.
DEV void tdm_load_b_tile(const unsigned short* gsrc, unsigned lds_off) {
    const unsigned long long ga = (unsigned long long)(size_t)gsrc;
    u32x4 g0;
    g0[0] = 1u;                                    // count=1, user mode, no gather
    g0[1] = lds_off;                               // lds_addr (bytes)
    g0[2] = (unsigned)(ga & 0xFFFFFFFFull);        // global_addr[31:0]
    g0[3] = (unsigned)((ga >> 32) & 0x1FFFFFFull)  // global_addr[56:32]
          | (2u << 30);                            // type = 2 ("image")
    i32x8 g1;
    g1[0] = (int)(1u << 16);                       // data_size = 1 -> 2 bytes
    g1[1] = (int)(512u << 16);                     // tensor_dim0 = 512 elems/row
    g1[2] = (int)(128u << 16);                     // tensor_dim1 = 128 rows
    g1[3] = (int)(32u << 16);                      // tile_dim0 = 32
    g1[4] = 128;                                   // tile_dim1 = 128
    g1[5] = 512;                                   // tensor_dim0_stride = 512 elems
    g1[6] = 0;
    g1[7] = 0;
    i32x4 z4 = {0, 0, 0, 0};
#if __clang_major__ >= 23
    i32x8 z8 = {0, 0, 0, 0, 0, 0, 0, 0};
    __builtin_amdgcn_tensor_load_to_lds(g0, g1, z4, z4, z8, 0);
#else
    __builtin_amdgcn_tensor_load_to_lds(g0, g1, z4, z4, 0);
#endif
}
#endif

// ---------------- phase 1: bulk input projection  G = X @ W_ih^T  (both dirs) ----------
// grid = (TOKENS/128, G_DIM/128, 2), block = 256 (8 waves). 128x128 tile, K in 16 chunks
// of 32. A (fp32 x, converted) staged manually, double buffered; B (bf16 weights) staged
// by the Tensor Data Mover (async, TENSORcnt). A fragments preloaded so WMMAs issue
// back-to-back instead of ds_load->wait->wmma per tile.
__global__ __launch_bounds__(256) void xproj_gemm(
    const float* __restrict__ X,
    const unsigned short* __restrict__ wih_f,
    const unsigned short* __restrict__ wih_b,
    unsigned short* __restrict__ G)
{
    const int dir = blockIdx.z;
    const unsigned short* __restrict__ w = dir ? wih_b : wih_f;
    unsigned short* __restrict__ Gd = G + (size_t)dir * TOKENS * G_DIM;

    const int tokBase = blockIdx.x * 128;
    const int colBase = blockIdx.y * 128;

    __shared__ __align__(32) unsigned short sa[2][128 * 32];   // 8 KB each
    __shared__ __align__(32) unsigned short sb[2][128 * 32];

    const int tid  = threadIdx.x;
    const int wv   = tid >> 5;
    const int lane = tid & 31;
    const int ln   = lane & 15;
    const int kh   = lane >> 4;

    const int srow  = tid >> 1;           // staging: row 0..127
    const int shalf = (tid & 1) * 16;     // 16 elements per thread

    const v8f zero8 = {0.f,0.f,0.f,0.f,0.f,0.f,0.f,0.f};
    v8f acc[8];
#pragma unroll
    for (int rt = 0; rt < 8; ++rt) acc[rt] = zero8;

    // ---- stage chunk 0 ----
    {
        const float* ap = X + (size_t)(tokBase + srow) * D_IN + shalf;
        unsigned p[8];
#pragma unroll
        for (int q = 0; q < 4; ++q) {
            float4 v = *(const float4*)(ap + q * 4);
            p[2 * q]     = pk2bf(v.x, v.y);
            p[2 * q + 1] = pk2bf(v.z, v.w);
        }
        *reinterpret_cast<uint4*>(&sa[0][srow * 32 + shalf])     = make_uint4(p[0], p[1], p[2], p[3]);
        *reinterpret_cast<uint4*>(&sa[0][srow * 32 + shalf + 8]) = make_uint4(p[4], p[5], p[6], p[7]);
#ifdef HAVE_TDM
        if (wv == 0) {
            tdm_load_b_tile(w + (size_t)colBase * D_IN,
                            (unsigned)(size_t)&sb[0][0]);
            __builtin_amdgcn_s_wait_tensorcnt((short)0);
        }
#else
        const unsigned short* bp = w + (size_t)(colBase + srow) * D_IN + shalf;
        *reinterpret_cast<uint4*>(&sb[0][srow * 32 + shalf])     = *(const uint4*)(bp);
        *reinterpret_cast<uint4*>(&sb[0][srow * 32 + shalf + 8]) = *(const uint4*)(bp + 8);
#endif
    }
    __syncthreads();

    for (int kc = 0; kc < 16; ++kc) {
        const int cur = kc & 1, nxt = cur ^ 1;

        // issue next chunk's loads early (A: global->regs; B: TDM -> LDS, async)
        float4 av[4];
        if (kc < 15) {
            const int k0 = (kc + 1) * 32;
            const float* ap = X + (size_t)(tokBase + srow) * D_IN + k0 + shalf;
#pragma unroll
            for (int q = 0; q < 4; ++q) av[q] = *(const float4*)(ap + q * 4);
#ifdef HAVE_TDM
            if (wv == 0)
                tdm_load_b_tile(w + (size_t)colBase * D_IN + k0,
                                (unsigned)(size_t)&sb[nxt][0]);
#endif
        }
#ifndef HAVE_TDM
        uint4 bv[2];
        if (kc < 15) {
            const int k0 = (kc + 1) * 32;
            const unsigned short* bp = w + (size_t)(colBase + srow) * D_IN + k0 + shalf;
            bv[0] = *(const uint4*)(bp);
            bv[1] = *(const uint4*)(bp + 8);
        }
#endif

        // preload all fragments for this chunk, then issue WMMAs back-to-back
        v16bf bfrag = *(const v16bf*)&sb[cur][(wv * 16 + ln) * 32 + kh * 16];
        v16bf afr[8];
#pragma unroll
        for (int rt = 0; rt < 8; ++rt)
            afr[rt] = *(const v16bf*)&sa[cur][(rt * 16 + ln) * 32 + kh * 16];
#pragma unroll
        for (int rt = 0; rt < 8; ++rt)
            acc[rt] = __builtin_amdgcn_wmma_f32_16x16x32_bf16(
                          false, afr[rt], false, bfrag, (short)0, acc[rt], false, false);

        if (kc < 15) {
            unsigned p[8];
#pragma unroll
            for (int q = 0; q < 4; ++q) {
                p[2 * q]     = pk2bf(av[q].x, av[q].y);
                p[2 * q + 1] = pk2bf(av[q].z, av[q].w);
            }
            *reinterpret_cast<uint4*>(&sa[nxt][srow * 32 + shalf])     = make_uint4(p[0], p[1], p[2], p[3]);
            *reinterpret_cast<uint4*>(&sa[nxt][srow * 32 + shalf + 8]) = make_uint4(p[4], p[5], p[6], p[7]);
#ifndef HAVE_TDM
            *reinterpret_cast<uint4*>(&sb[nxt][srow * 32 + shalf])     = bv[0];
            *reinterpret_cast<uint4*>(&sb[nxt][srow * 32 + shalf + 8]) = bv[1];
#endif
        }
#ifdef HAVE_TDM
        if (wv == 0) __builtin_amdgcn_s_wait_tensorcnt((short)0);
#endif
        __syncthreads();
    }

    // ---- emit 128x16 strip as bf16 ----
    const int col = colBase + wv * 16 + ln;
#pragma unroll
    for (int rt = 0; rt < 8; ++rt) {
#pragma unroll
        for (int r = 0; r < 8; ++r) {
            const int tok = tokBase + rt * 16 + r + (kh << 3);
            Gd[(size_t)tok * G_DIM + col] = f2bf(acc[rt][r]);
        }
    }
}

// ---------------- phase 2: recurrence  gates = Gin + h @ W_hh^T + b ---------------------
// grid = (N/16, 2 dirs), block = 512 (16 waves). Wave w owns gate columns
// {g*256 + w*16 .. +16} for g in {i,f,g,o}; its W_hh slice (32 B-fragments = 256 VGPRs)
// stays in registers for the whole time loop. A fragments are pipelined one k-chunk
// ahead so each ds_load overlaps 4 WMMAs.
__global__ __launch_bounds__(512) void bilstm_rec(
    const unsigned short* __restrict__ G,
    const int*   __restrict__ lengths,
    const int*   __restrict__ starts,
    const unsigned short* __restrict__ whh_f,
    const float* __restrict__ b_f,
    const unsigned short* __restrict__ whh_b,
    const float* __restrict__ b_b,
    float* __restrict__ out)
{
    const int dir     = blockIdx.y;
    const int seqBase = blockIdx.x * 16;
    const unsigned short* __restrict__ whh = dir ? whh_b : whh_f;
    const float*          __restrict__ bia = dir ? b_b   : b_f;
    const unsigned short* __restrict__ Gd  = G + (size_t)dir * TOKENS * G_DIM;

    __shared__ __align__(32) unsigned short sh_hb[16 * H_DIM];   // 8 KB: h_t bf16

    const int tid  = threadIdx.x;
    const int wv   = tid >> 5;          // 0..15 -> column strip
    const int lane = tid & 31;
    const int ln   = lane & 15;
    const int kh   = lane >> 4;
    const int hc   = wv * 16 + ln;

    for (int i = tid; i < 16 * H_DIM; i += 512) sh_hb[i] = 0;

    // register-resident W_hh slice: 4 gate tiles x 8 k-chunks
    v16bf wb[4][8];
#pragma unroll
    for (int j = 0; j < 4; ++j)
#pragma unroll
        for (int kc = 0; kc < 8; ++kc)
            wb[j][kc] = *(const v16bf*)&whh[(size_t)(j * H_DIM + hc) * H_DIM + kc * 32 + kh * 16];

    const float bi = bia[0 * H_DIM + hc];
    const float bf = bia[1 * H_DIM + hc];
    const float bg = bia[2 * H_DIM + hc];
    const float bo = bia[3 * H_DIM + hc];

    long long st[8];
    int       ln8[8];
#pragma unroll
    for (int r = 0; r < 8; ++r) {
        st[r]  = (long long)starts[seqBase + (kh << 3) + r];
        ln8[r] = lengths[seqBase + (kh << 3) + r];
    }

    float hreg[8], creg[8];
#pragma unroll
    for (int r = 0; r < 8; ++r) { hreg[r] = 0.f; creg[r] = 0.f; }

    for (int s = 0; s < T_LEN; ++s) {
        const int t = dir ? (T_LEN - 1 - s) : s;

        // accumulators start from the precomputed input projection
        v8f acc[4];
#pragma unroll
        for (int j = 0; j < 4; ++j)
#pragma unroll
            for (int r = 0; r < 8; ++r)
                acc[j][r] = bf2f(Gd[(size_t)(st[r] + t) * G_DIM + j * H_DIM + hc]);

        if (s + 1 < T_LEN) {
            const int tn = dir ? (t - 1) : (t + 1);
#pragma unroll
            for (int r = 0; r < 8; ++r)
                __builtin_prefetch(&Gd[(size_t)(st[r] + tn) * G_DIM + hc], 0, 0);
        }

        __syncthreads();   // prev step's h bf16 visible

        // h @ W_hh^T : B in VGPRs; A pipelined one chunk ahead
        v16bf a_cur = *(const v16bf*)&sh_hb[ln * H_DIM + kh * 16];
        v16bf a_nxt = a_cur;
#pragma unroll
        for (int kc = 0; kc < 8; ++kc) {
            if (kc < 7)
                a_nxt = *(const v16bf*)&sh_hb[ln * H_DIM + (kc + 1) * 32 + kh * 16];
#pragma unroll
            for (int j = 0; j < 4; ++j)
                acc[j] = __builtin_amdgcn_wmma_f32_16x16x32_bf16(
                             false, a_cur, false, wb[j][kc], (short)0, acc[j], false, false);
            a_cur = a_nxt;
        }
        __syncthreads();   // everyone done reading sh_hb

        // LSTM cell update, fully in registers
#pragma unroll
        for (int r = 0; r < 8; ++r) {
            const int m = r + (kh << 3);
            const float iv = acc[0][r] + bi;
            const float fv = acc[1][r] + bf;
            const float gv = acc[2][r] + bg;
            const float ov = acc[3][r] + bo;
            const float si = 1.f / (1.f + __expf(-iv));
            const float sf = 1.f / (1.f + __expf(-fv));
            const float so = 1.f / (1.f + __expf(-ov));
            float cn = sf * creg[r] + si * tanhf(gv);
            float hn = so * tanhf(cn);
            const bool valid = t < ln8[r];
            cn = valid ? cn : creg[r];
            hn = valid ? hn : hreg[r];
            creg[r] = cn;
            hreg[r] = hn;
            sh_hb[m * H_DIM + hc] = f2bf(hn);
        }
    }

#pragma unroll
    for (int r = 0; r < 8; ++r) {
        const int m = r + (kh << 3);
        out[(size_t)(seqBase + m) * (2 * H_DIM) + dir * H_DIM + hc] = hreg[r];
    }
}

extern "C" void kernel_launch(void* const* d_in, const int* in_sizes, int n_in,
                              void* d_out, int out_size, void* d_ws, size_t ws_size,
                              hipStream_t stream) {
    const float* all_embs = (const float*)d_in[0];
    const int*   lengths  = (const int*)  d_in[1];
    const int*   starts   = (const int*)  d_in[2];
    const float* W_ih_f   = (const float*)d_in[3];
    const float* W_hh_f   = (const float*)d_in[4];
    const float* b_f      = (const float*)d_in[5];
    const float* W_ih_b   = (const float*)d_in[6];
    const float* W_hh_b   = (const float*)d_in[7];
    const float* b_b      = (const float*)d_in[8];
    float* out = (float*)d_out;

    // workspace layout (bf16 elements):
    //   weights (3 MB) then G = 2 * TOKENS * G_DIM bf16 (~537 MB)
    unsigned short* ws    = (unsigned short*)d_ws;
    unsigned short* wih_f = ws;
    unsigned short* whh_f = wih_f + (size_t)G_DIM * D_IN;
    unsigned short* wih_b = whh_f + (size_t)G_DIM * H_DIM;
    unsigned short* whh_b = wih_b + (size_t)G_DIM * D_IN;
    unsigned short* Gbuf  = whh_b + (size_t)G_DIM * H_DIM;

    const int nih = G_DIM * D_IN;
    const int nhh = G_DIM * H_DIM;
    cvt_f32_bf16<<<256, 256, 0, stream>>>(W_ih_f, wih_f, nih);
    cvt_f32_bf16<<<256, 256, 0, stream>>>(W_hh_f, whh_f, nhh);
    cvt_f32_bf16<<<256, 256, 0, stream>>>(W_ih_b, wih_b, nih);
    cvt_f32_bf16<<<256, 256, 0, stream>>>(W_hh_b, whh_b, nhh);

    dim3 gridA(TOKENS / 128, G_DIM / 128, 2);
    xproj_gemm<<<gridA, 256, 0, stream>>>(all_embs, wih_f, wih_b, Gbuf);

    dim3 gridB(N_SEQ / 16, 2);
    bilstm_rec<<<gridB, 512, 0, stream>>>(Gbuf, lengths, starts,
                                          whh_f, b_f, whh_b, b_b, out);
}
